// QLinear_9423158247731
// MI455X (gfx1250) — matrix operand order repaired and checked by reading, not compiled, not run on hardware
//
#include <hip/hip_runtime.h>

typedef __bf16 bf16_t;
typedef __attribute__((ext_vector_type(16))) __bf16 v16bf;
typedef __attribute__((ext_vector_type(8)))  __bf16 v8bf;
typedef __attribute__((ext_vector_type(4)))  __bf16 v4bf;
typedef __attribute__((ext_vector_type(2)))  __bf16 v2bf;
typedef __attribute__((ext_vector_type(8)))  float  v8f;

static constexpr int O_DIM   = 4096;       // out features
static constexpr int I_DIM   = 4096;       // in features
static constexpr int NGROUPS = 32;         // I / 128
static constexpr int QCOLS   = I_DIM / 2;  // packed byte columns

static constexpr int MT  = 128;  // M tile per block
static constexpr int NT  = 256;  // N tile per block
static constexpr int KT  = 64;   // K tile per block
static constexpr int LDX = 72;   // padded LDS row stride (bf16 elems; 144B rows, 16B & 8B aligned)

__global__ __launch_bounds__(256, 1)
void qlinear_wmma_bf16(const float* __restrict__ x,
                       const int*   __restrict__ qweight,
                       const float* __restrict__ scale,
                       const float* __restrict__ zero,
                       const float* __restrict__ bias,
                       const int*   __restrict__ perm,
                       float*       __restrict__ out)
{
    __shared__ __align__(16) bf16_t xs[MT][LDX];   // x tile, bf16, [m][k]
    __shared__ __align__(16) bf16_t ws[NT][LDX];   // dequant weight tile, bf16, [n][k]
    __shared__ float2 szbuf[NT];                   // per-row (scale, scale*zero)

    const int tid  = threadIdx.x;
    const int lane = tid & 31;
    const int ln   = lane & 15;   // lane-low: row/col inside 16x16 tile
    const int hi   = lane >> 4;   // lane-high: K-half / M-half selector
    const int wave = tid >> 5;
    const int mBase = (wave & 1) * 64;   // 2 wave-rows of 64
    const int nBase = (wave >> 1) * 64;  // 4 wave-cols of 64

    const int n0 = blockIdx.x * NT;
    const int m0 = blockIdx.y * MT;

    // x-fill mapping: pairs of columns
    const int xc = (tid & 31) * 2;   // column pair base (0..62)
    const int xr = tid >> 5;         // row phase (0..7), rows xr + 8*i
    // w-fill mapping: quads of columns
    const int wc = (tid & 15) * 4;   // column quad base (0..60)
    const int wr = tid >> 4;         // row phase (0..15), rows wr + 16*i

    v8f acc[4][4];
#pragma unroll
    for (int mt = 0; mt < 4; ++mt)
#pragma unroll
        for (int nt = 0; nt < 4; ++nt)
            acc[mt][nt] = (v8f){0.f, 0.f, 0.f, 0.f, 0.f, 0.f, 0.f, 0.f};

    for (int k0 = 0; k0 < I_DIM; k0 += KT) {
        const bool hiNib = (k0 < QCOLS);   // first half of I -> high nibble
        const int  shift = hiNib ? 4 : 0;
        const int  jc    = (hiNib ? k0 : k0 - QCOLS) + wc;  // packed byte column

        // Stage (scale, scale*zero) once per quant group; one row per thread, branch-free.
        if ((k0 & 127) == 0) {
            const int g = k0 >> 7;
            const float s = scale[(size_t)(n0 + tid) * NGROUPS + g];
            const float z = zero [(size_t)(n0 + tid) * NGROUPS + g];
            szbuf[tid] = make_float2(s, s * z);
        }
        __syncthreads();  // fences prior WMMA reads of xs/ws AND szbuf writes

        // ---- fill x tile: two f32 gathers -> one packed bf16x2 LDS store ----
        const int2 pp = *(const int2*)&perm[k0 + xc];
#pragma unroll
        for (int i = 0; i < 16; ++i) {
            const int r = xr + 8 * i;
            const float f0 = x[(size_t)(m0 + r) * I_DIM + pp.x];
            const float f1 = x[(size_t)(m0 + r) * I_DIM + pp.y];
            v2bf p;
            p.x = (bf16_t)f0;
            p.y = (bf16_t)f1;
            *(v2bf*)&xs[r][xc] = p;
        }
        // ---- fill weight tile: b128 load of 4 packed bytes -> 4 dequants -> b64 store ----
#pragma unroll
        for (int i = 0; i < 16; ++i) {
            const int r = wr + 16 * i;
            const int4 v = *(const int4*)&qweight[(size_t)(n0 + r) * QCOLS + jc];
            const float2 sz = szbuf[r];
            const float q0 = (float)((v.x >> shift) & 15);
            const float q1 = (float)((v.y >> shift) & 15);
            const float q2 = (float)((v.z >> shift) & 15);
            const float q3 = (float)((v.w >> shift) & 15);
            v4bf wv;
            wv.x = (bf16_t)fmaf(q0, sz.x, -sz.y);
            wv.y = (bf16_t)fmaf(q1, sz.x, -sz.y);
            wv.z = (bf16_t)fmaf(q2, sz.x, -sz.y);
            wv.w = (bf16_t)fmaf(q3, sz.x, -sz.y);
            *(v4bf*)&ws[r][wc] = wv;
        }

        // ---- prefetch next K-tile into cache while this tile's WMMAs run ----
        // (global_prefetch_b8: no VGPR result, no counter; pure latency hider.
        //  x prefetch uses the un-permuted column — a hint, exact for identity perm.)
        if (k0 + KT < I_DIM) {
            const int kn   = k0 + KT;
            const int jb2  = (kn < QCOLS) ? kn : kn - QCOLS;
            // next x tile: 128 rows x 256B  -> 256 x 128B lines, one per thread
            __builtin_prefetch(&x[(size_t)(m0 + (tid >> 1)) * I_DIM + kn + (tid & 1) * 32], 0, 3);
            // next qweight tile: 256 rows x 256B -> 512 x 128B lines, two per thread
            __builtin_prefetch(&qweight[(size_t)(n0 + tid) * QCOLS + jb2], 0, 3);
            __builtin_prefetch(&qweight[(size_t)(n0 + tid) * QCOLS + jb2 + 32], 0, 3);
        }
        __syncthreads();

        // ---- compute: two K=32 WMMA steps over the K=64 tile ----
#pragma unroll
        for (int kb = 0; kb < KT; kb += 32) {
            v16bf a[4], b[4];
            // A fragment: 16x32 bf16; lane<16 holds K {0..7,16..23}, lane>=16 {8..15,24..31}
#pragma unroll
            for (int mt = 0; mt < 4; ++mt) {
                const bf16_t* row = &xs[mBase + mt * 16 + ln][0];
                v8bf lo  = *(const v8bf*)(row + kb + hi * 8);
                v8bf hi8 = *(const v8bf*)(row + kb + 16 + hi * 8);
                a[mt] = __builtin_shufflevector(lo, hi8,
                        0, 1, 2, 3, 4, 5, 6, 7, 8, 9, 10, 11, 12, 13, 14, 15);
            }
            // B fragment: 32x16 bf16; lane<16 = col n with K {0..7,16..23}, lane>=16 {8..15,24..31}
#pragma unroll
            for (int nt = 0; nt < 4; ++nt) {
                const bf16_t* row = &ws[nBase + nt * 16 + ln][0];
                v8bf lo  = *(const v8bf*)(row + kb + hi * 8);
                v8bf hi8 = *(const v8bf*)(row + kb + 16 + hi * 8);
                b[nt] = __builtin_shufflevector(lo, hi8,
                        0, 1, 2, 3, 4, 5, 6, 7, 8, 9, 10, 11, 12, 13, 14, 15);
            }
#pragma unroll
            for (int mt = 0; mt < 4; ++mt)
#pragma unroll
                for (int nt = 0; nt < 4; ++nt)
                    acc[mt][nt] = __builtin_amdgcn_wmma_f32_16x16x32_bf16(
                        false, a[mt], false, b[nt],
                        (short)0, acc[mt][nt], false, false);
        }
    }

    // ---- epilogue: bias add + store (C/D layout: VGPR j = row, lane = col) ----
#pragma unroll
    for (int nt = 0; nt < 4; ++nt) {
        const int n  = n0 + nBase + nt * 16 + ln;
        const float bv = bias[n];
#pragma unroll
        for (int mt = 0; mt < 4; ++mt) {
            const int mrow = m0 + mBase + mt * 16 + hi * 8;
#pragma unroll
            for (int j = 0; j < 8; ++j)
                out[(size_t)(mrow + j) * O_DIM + n] = acc[mt][nt][j] + bv;
        }
    }
}

extern "C" void kernel_launch(void* const* d_in, const int* in_sizes, int n_in,
                              void* d_out, int out_size, void* d_ws, size_t ws_size,
                              hipStream_t stream) {
    const float* x       = (const float*)d_in[0];
    const int*   qweight = (const int*)  d_in[1];
    const float* scale   = (const float*)d_in[2];
    const float* zero    = (const float*)d_in[3];
    const float* bias    = (const float*)d_in[4];
    const int*   perm    = (const int*)  d_in[5];
    float*       out     = (float*)d_out;

    const int M = in_sizes[0] / I_DIM;  // 8192 rows
    dim3 grid(O_DIM / NT, M / MT);      // (16, 64)
    dim3 block(256);                    // 8 wave32 waves
    hipLaunchKernelGGL(qlinear_wmma_bf16, grid, block, 0, stream,
                       x, qweight, scale, zero, bias, perm, out);
}
